// BiCrossAttention_53695681134600
// MI455X (gfx1250) — compile-verified, matrix operand order checked by reading, not run on hardware
//
#include <hip/hip_runtime.h>
#include <stdint.h>

// ---------------- types ----------------
typedef __attribute__((ext_vector_type(16))) __bf16 bf16x16;
typedef __attribute__((ext_vector_type(8)))  float  f32x8;

#define BM 128
#define BN 128
#define BK 32
#define LDSK 40   // padded K-stride (elements); 80B rows keep 16B alignment, kill bank conflicts

// RNE f32->bf16 (inputs are finite: activations / Xavier weights / softmax probs)
__device__ __forceinline__ unsigned short f32_to_bf16(float f) {
    union { float f; uint32_t u; } c; c.f = f;
    uint32_t u = c.u;
    return (unsigned short)((u + 0x7FFFu + ((u >> 16) & 1u)) >> 16);
}
__device__ __forceinline__ uint32_t pack2_bf16(float a, float b) {
    return (uint32_t)f32_to_bf16(a) | ((uint32_t)f32_to_bf16(b) << 16);
}

// ---------------- elementwise fp32 -> bf16 ----------------
__global__ __launch_bounds__(256) void cvt_f32_bf16(const float* __restrict__ src,
                                                    unsigned short* __restrict__ dst, int n) {
    int i = (blockIdx.x * 256 + threadIdx.x) * 4;
    if (i + 3 < n) {
        float4 f = *(const float4*)(src + i);
        uint2 o;
        o.x = pack2_bf16(f.x, f.y);
        o.y = pack2_bf16(f.z, f.w);
        *(uint2*)(dst + i) = o;
    }
}

// ---------------- double-buffered tiled WMMA GEMM ----------------
// C[M,N] = scale * A[M,K] * B(logical [K,N])
//  A_F32  : A is fp32 (converted to bf16 on the way into LDS), else A is bf16
//  B_TRANS: B memory layout is [K][N] row-major (transposed into LDS),
//           else B memory layout is [N][K] row-major (direct copy; K-matrix for Q.K^T)
//  C_BF16 : store bf16, else fp32
template <bool A_F32, bool B_TRANS, bool C_BF16>
__global__ __launch_bounds__(256)
void gemm_wmma(const void* __restrict__ Av, const unsigned short* __restrict__ Bv,
               void* __restrict__ Cv, int M, int N, int K,
               long long sA, long long sB, long long sC, float scale)
{
    __shared__ __align__(16) unsigned short As[2][BM * LDSK];
    __shared__ __align__(16) unsigned short Bs[2][BN * LDSK];

    const int tid = threadIdx.x;
    const int z   = blockIdx.z;
    const int n0  = blockIdx.x * BN;
    const int m0  = blockIdx.y * BM;

    const float*          Af = (const float*)Av + (long long)z * sA;
    const unsigned short* Ah = (const unsigned short*)Av + (long long)z * sA;
    const unsigned short* B  = Bv + (long long)z * sB;

    const int wave = tid >> 5;
    const int lane = tid & 31;
    const int wm   = (wave >> 2) * 64;   // 0 or 64
    const int wn   = (wave & 3) * 32;    // 0..96

    f32x8 acc[4][2];
#pragma unroll
    for (int i = 0; i < 4; ++i)
#pragma unroll
        for (int j = 0; j < 2; ++j) acc[i][j] = {};

    const int arow  = tid >> 1;          // 0..127 (A / direct-B row)
    const int ahalf = (tid & 1) * 16;    // k offset 0 or 16
    const int bkrow = tid >> 3;          // 0..31   (transposed-B source row)
    const int bnoff = (tid & 7) * 16;    // 0..112  (transposed-B col group)

    // ---- staging registers for the in-flight global tile ----
    float4 fa[4];   // A staging when A_F32
    uint4  ua[2];   // A staging when bf16
    uint4  ub[2];   // B staging

    auto gload = [&](int kt) {
        const int kbase = kt * BK;
        if constexpr (A_F32) {
            const float4* s4 = (const float4*)(Af + (long long)(m0 + arow) * K + kbase + ahalf);
            fa[0] = s4[0]; fa[1] = s4[1]; fa[2] = s4[2]; fa[3] = s4[3];
        } else {
            const uint4* s4 = (const uint4*)(Ah + (long long)(m0 + arow) * K + kbase + ahalf);
            ua[0] = s4[0]; ua[1] = s4[1];
        }
        if constexpr (B_TRANS) {
            const uint4* s4 = (const uint4*)(B + (long long)(kbase + bkrow) * N + n0 + bnoff);
            ub[0] = s4[0]; ub[1] = s4[1];
        } else {
            const uint4* s4 = (const uint4*)(B + (long long)(n0 + arow) * K + kbase + ahalf);
            ub[0] = s4[0]; ub[1] = s4[1];
        }
    };

    auto lstore = [&](int buf) {
        if constexpr (A_F32) {
            uint4 q0, q1;
            q0.x = pack2_bf16(fa[0].x, fa[0].y); q0.y = pack2_bf16(fa[0].z, fa[0].w);
            q0.z = pack2_bf16(fa[1].x, fa[1].y); q0.w = pack2_bf16(fa[1].z, fa[1].w);
            q1.x = pack2_bf16(fa[2].x, fa[2].y); q1.y = pack2_bf16(fa[2].z, fa[2].w);
            q1.z = pack2_bf16(fa[3].x, fa[3].y); q1.w = pack2_bf16(fa[3].z, fa[3].w);
            uint4* dst = (uint4*)&As[buf][arow * LDSK + ahalf];
            dst[0] = q0; dst[1] = q1;
        } else {
            uint4* dst = (uint4*)&As[buf][arow * LDSK + ahalf];
            dst[0] = ua[0]; dst[1] = ua[1];
        }
        if constexpr (B_TRANS) {
            const unsigned short* v0 = (const unsigned short*)&ub[0];
            const unsigned short* v1 = (const unsigned short*)&ub[1];
#pragma unroll
            for (int j = 0; j < 8; ++j) Bs[buf][(bnoff + j) * LDSK + bkrow] = v0[j];
#pragma unroll
            for (int j = 0; j < 8; ++j) Bs[buf][(bnoff + 8 + j) * LDSK + bkrow] = v1[j];
        } else {
            uint4* dst = (uint4*)&Bs[buf][arow * LDSK + ahalf];
            dst[0] = ub[0]; dst[1] = ub[1];
        }
    };

    auto compute = [&](int buf) {
        const int frow = lane & 15;
        const int fk   = (lane >> 4) * 8;
        union Frag { bf16x16 v; uint4 q[2]; } afr[4], bfr[2];
#pragma unroll
        for (int i = 0; i < 4; ++i) {
            const unsigned short* p = &As[buf][(wm + i * 16 + frow) * LDSK + fk];
            afr[i].q[0] = *(const uint4*)p;
            afr[i].q[1] = *(const uint4*)(p + 16);
        }
#pragma unroll
        for (int j = 0; j < 2; ++j) {
            const unsigned short* p = &Bs[buf][(wn + j * 16 + frow) * LDSK + fk];
            bfr[j].q[0] = *(const uint4*)p;
            bfr[j].q[1] = *(const uint4*)(p + 16);
        }
#pragma unroll
        for (int i = 0; i < 4; ++i)
#pragma unroll
            for (int j = 0; j < 2; ++j)
                acc[i][j] = __builtin_amdgcn_wmma_f32_16x16x32_bf16(
                    false, afr[i].v, false, bfr[j].v, (short)0, acc[i][j], false, false);
    };

    // ---- ping-pong main loop: one barrier per K-step ----
    const int ksteps = K / BK;
    gload(0);
    lstore(0);
    __syncthreads();
    for (int kt = 0; kt < ksteps; ++kt) {
        const int cur = kt & 1;
        if (kt + 1 < ksteps) gload(kt + 1);   // HBM/L2 latency hides under WMMA below
        compute(cur);
        if (kt + 1 < ksteps) {
            lstore(cur ^ 1);                  // other buffer: no readers until next barrier
            __syncthreads();
        }
    }

    // ---- epilogue (ISA 32-bit 16x16 C/D layout) ----
    const int coln = lane & 15;
    const int rh   = (lane >> 4) * 8;
#pragma unroll
    for (int i = 0; i < 4; ++i) {
#pragma unroll
        for (int j = 0; j < 2; ++j) {
            const long long col = n0 + wn + j * 16 + coln;
#pragma unroll
            for (int r = 0; r < 8; ++r) {
                const long long row = m0 + wm + i * 16 + rh + r;
                float v = acc[i][j][r] * scale;
                if constexpr (C_BF16)
                    ((unsigned short*)Cv)[(long long)z * sC + row * N + col] = f32_to_bf16(v);
                else
                    ((float*)Cv)[(long long)z * sC + row * N + col] = v;
            }
        }
    }
}

// ---------------- row softmax: fp32 scores -> bf16 probabilities ----------------
__global__ __launch_bounds__(256)
void softmax_rows(const float* __restrict__ S, unsigned short* __restrict__ P, int len)
{
    __shared__ float red[256];
    const long long row = blockIdx.x;
    const float* s = S + row * (long long)len;
    unsigned short* p = P + row * (long long)len;
    const int tid = threadIdx.x;

    float loc[8];                      // len == 2048, 8 per thread
    float m = -3.4e38f;
#pragma unroll
    for (int i = 0; i < 8; ++i) { loc[i] = s[tid + i * 256]; m = fmaxf(m, loc[i]); }
    red[tid] = m; __syncthreads();
    for (int off = 128; off > 0; off >>= 1) {
        if (tid < off) red[tid] = fmaxf(red[tid], red[tid + off]);
        __syncthreads();
    }
    m = red[0]; __syncthreads();

    float sum = 0.f;
#pragma unroll
    for (int i = 0; i < 8; ++i) { loc[i] = __expf(loc[i] - m); sum += loc[i]; }
    red[tid] = sum; __syncthreads();
    for (int off = 128; off > 0; off >>= 1) {
        if (tid < off) red[tid] += red[tid + off];
        __syncthreads();
    }
    const float inv = 1.0f / red[0];
#pragma unroll
    for (int i = 0; i < 8; ++i) p[tid + i * 256] = f32_to_bf16(loc[i] * inv);
}

// ---------------- host orchestration ----------------
extern "C" void kernel_launch(void* const* d_in, const int* in_sizes, int n_in,
                              void* d_out, int out_size, void* d_ws, size_t ws_size,
                              hipStream_t stream)
{
    const int B = 8, S = 2048, D = 1024;
    const long long MS = (long long)B * S;                 // 16384 rows flattened

    char* ws = (char*)d_ws;
    size_t off = 0;
    // bf16 weights (dict order: Wq1, Wk1, Wv1, Wq2, Wk2, Wv2)
    unsigned short* wb[6];
    for (int i = 0; i < 6; ++i) { wb[i] = (unsigned short*)(ws + off); off += (size_t)D * D * 2; }
    // bf16 projections
    unsigned short* Qa = (unsigned short*)(ws + off); off += (size_t)MS * D * 2;  // x1*Wq1
    unsigned short* Ka = (unsigned short*)(ws + off); off += (size_t)MS * D * 2;  // x2*Wk2
    unsigned short* Va = (unsigned short*)(ws + off); off += (size_t)MS * D * 2;  // x2*Wv2
    unsigned short* Qb = (unsigned short*)(ws + off); off += (size_t)MS * D * 2;  // x2*Wq2
    unsigned short* Kb = (unsigned short*)(ws + off); off += (size_t)MS * D * 2;  // x1*Wk1
    unsigned short* Vb = (unsigned short*)(ws + off); off += (size_t)MS * D * 2;  // x1*Wv1
    float*          Sc = (float*)(ws + off);          off += (size_t)B * S * S * 4;
    unsigned short* Pm = (unsigned short*)(ws + off); off += (size_t)B * S * S * 2;

    const float* x1 = (const float*)d_in[0];
    const float* x2 = (const float*)d_in[1];

    // 1) weights -> bf16
    for (int i = 0; i < 6; ++i)
        cvt_f32_bf16<<<(D * D / 4 + 255) / 256, 256, 0, stream>>>((const float*)d_in[2 + i], wb[i], D * D);

    // 2) projections: [16384,1024] x [1024,1024] -> bf16
    dim3 gp(D / BN, (unsigned)(MS / BM), 1);
    gemm_wmma<true, true, true><<<gp, 256, 0, stream>>>(x1, wb[0], Qa, (int)MS, D, D, 0, 0, 0, 1.0f); // Q1=x1*Wq1
    gemm_wmma<true, true, true><<<gp, 256, 0, stream>>>(x1, wb[1], Kb, (int)MS, D, D, 0, 0, 0, 1.0f); // K =x1*Wk1
    gemm_wmma<true, true, true><<<gp, 256, 0, stream>>>(x1, wb[2], Vb, (int)MS, D, D, 0, 0, 0, 1.0f); // V =x1*Wv1
    gemm_wmma<true, true, true><<<gp, 256, 0, stream>>>(x2, wb[3], Qb, (int)MS, D, D, 0, 0, 0, 1.0f); // Q2=x2*Wq2
    gemm_wmma<true, true, true><<<gp, 256, 0, stream>>>(x2, wb[4], Ka, (int)MS, D, D, 0, 0, 0, 1.0f); // K =x2*Wk2
    gemm_wmma<true, true, true><<<gp, 256, 0, stream>>>(x2, wb[5], Va, (int)MS, D, D, 0, 0, 0, 1.0f); // V =x2*Wv2

    float* out1 = (float*)d_out;                                   // context_vec_1
    float* out2 = (float*)d_out + (long long)B * S * (long long)D; // context_vec_2

    dim3 gs(S / BN, S / BM, B);   // scores: per-batch [2048 x 2048], K=1024
    dim3 go(D / BN, S / BM, B);   // out:    per-batch [2048 x 1024], K=2048
    const float invsqrt = 0.03125f; // 1/sqrt(1024)

    // direction A: Q from x1, K/V from x2 -> context_vec_2
    gemm_wmma<false, false, false><<<gs, 256, 0, stream>>>(Qa, Ka, Sc, S, S, D,
        (long long)S * D, (long long)S * D, (long long)S * S, invsqrt);
    softmax_rows<<<(unsigned)(B * S), 256, 0, stream>>>(Sc, Pm, S);
    gemm_wmma<false, true, false><<<go, 256, 0, stream>>>(Pm, Va, out2, S, D, S,
        (long long)S * S, (long long)S * D, (long long)S * D, 1.0f);

    // direction B: Q from x2, K/V from x1 -> context_vec_1
    gemm_wmma<false, false, false><<<gs, 256, 0, stream>>>(Qb, Kb, Sc, S, S, D,
        (long long)S * D, (long long)S * D, (long long)S * S, invsqrt);
    softmax_rows<<<(unsigned)(B * S), 256, 0, stream>>>(Sc, Pm, S);
    gemm_wmma<false, true, false><<<go, 256, 0, stream>>>(Pm, Vb, out1, S, D, S,
        (long long)S * S, (long long)S * D, (long long)S * D, 1.0f);

    (void)in_sizes; (void)n_in; (void)out_size; (void)ws_size;
}
